// SeqdistModel_90898687852760
// MI455X (gfx1250) — matrix operand order repaired and checked by reading, not compiled, not run on hardware
//
#include <hip/hip_runtime.h>

typedef _Float16 v16h __attribute__((ext_vector_type(16)));
typedef _Float16 v8h  __attribute__((ext_vector_type(8)));
typedef float    v8f  __attribute__((ext_vector_type(8)));
typedef int      vi4  __attribute__((vector_size(16)));   // matches builtin pointee

#define T_STEPS 800
#define NBATCH  32
#define FEAT    320
#define KW      19
#define STRIDE  5
#define LIN     4000
#define CSTATES 1024
#define NZ      5
#define COUT    5120            // C*NZ
#define MROWS   (T_STEPS * NBATCH)   // 25600

// ---- CDNA5 feature probes (guarded: absent builtin -> portable fallback) ----
#if defined(__has_builtin)
#if __has_builtin(__builtin_amdgcn_global_load_async_to_lds_b128) && \
    __has_builtin(__builtin_amdgcn_s_wait_asynccnt)
#define HAS_ASYNC_LDS 1
#endif
#endif

#define AS_GLOBAL __attribute__((address_space(1)))
#define AS_LDS    __attribute__((address_space(3)))

__device__ __forceinline__ float fast_tanh(float x) {
#if defined(__has_builtin) && __has_builtin(__builtin_amdgcn_tanhf)
    return __builtin_amdgcn_tanhf(x);        // V_TANH_F32 (gfx1250 trans op)
#elif defined(__has_builtin) && __has_builtin(__builtin_amdgcn_tanh_f32)
    return __builtin_amdgcn_tanh_f32(x);
#else
    return tanhf(x);
#endif
}

// ---------------- lin_w f32 [320][5120] -> Bt f16 [5120][320] ----------------
__global__ void k_wcvt(const float* __restrict__ lw, _Float16* __restrict__ Bt) {
    int i = blockIdx.x * 256 + threadIdx.x;          // over 320*5120
    if (i >= FEAT * COUT) return;
    int f = i / COUT, c = i % COUT;
    Bt[(size_t)c * FEAT + f] = (_Float16)lw[i];
}

// ---------------- conv1d(stride 5, SAME) + tanh -> feat f16 [25600][320] ----
__global__ void k_conv(const float* __restrict__ x, const float* __restrict__ w,
                       const float* __restrict__ b, _Float16* __restrict__ feat) {
    int gid = blockIdx.x * 256 + threadIdx.x;        // over 25600*320
    if (gid >= MROWS * FEAT) return;
    int f = gid % FEAT;
    int r = gid / FEAT;                              // r = t*32 + n
    int n = r & (NBATCH - 1);
    int t = r >> 5;
    const float* xp = x + (size_t)n * LIN;
    float acc = b[f];
    int p0 = t * STRIDE - 7;                         // SAME pad: left 7
#pragma unroll
    for (int i = 0; i < KW; ++i) {
        int p = p0 + i;
        float xv = (p >= 0 && p < LIN) ? xp[p] : 0.0f;
        acc = fmaf(xv, w[f * KW + i], acc);
    }
    feat[(size_t)r * FEAT + f] = (_Float16)fast_tanh(acc);
}

// ---------------- WMMA GEMM + fused 5*tanh(.+bias) epilogue -----------------
#define BM 128
#define BN 128
#define BK 32
#define KT (FEAT / BK)   // 10 k-tiles
#define LDP 40           // padded LDS row (f16): conflict-free fragment gathers

__device__ __forceinline__ v16h frag16(const _Float16* p0, const _Float16* p1) {
    v8h lo = *reinterpret_cast<const v8h*>(p0);
    v8h hi = *reinterpret_cast<const v8h*>(p1);
    return __builtin_shufflevector(lo, hi, 0,1,2,3,4,5,6,7,8,9,10,11,12,13,14,15);
}

__global__ __launch_bounds__(256)
void k_gemm(const _Float16* __restrict__ A,  // [25600][320] f16
            const _Float16* __restrict__ Bt, // [5120][320]  f16 (N-major)
            const float* __restrict__ bias,  // [5120]
            float* __restrict__ out) {       // [25600][5120] f32
    const int tid  = threadIdx.x;
    const int lane = tid & 31;
    const int wave = tid >> 5;               // 8 waves
    const int wm   = wave >> 2;              // 0..1  (64 M-rows each)
    const int wn   = wave & 3;               // 0..3  (32 N-cols each)
    const int lm   = lane & 15;
    const int lh   = lane >> 4;

    const int mBase = blockIdx.y * BM;
    const int nBase = blockIdx.x * BN;

    const int lrow  = tid >> 1;              // 0..127
    const int lhalf = tid & 1;               // 16-f16 chunk select

    v8f acc[4][2];
#pragma unroll
    for (int i = 0; i < 4; ++i)
#pragma unroll
        for (int j = 0; j < 2; ++j)
#pragma unroll
            for (int e = 0; e < 8; ++e) acc[i][j][e] = 0.0f;

#if HAS_ASYNC_LDS
    // ---- double-buffered, VGPR-free tile movement via GLOBAL_LOAD_ASYNC_TO_LDS
    __shared__ __align__(16) _Float16 Ash[2][BM * LDP];
    __shared__ __align__(16) _Float16 Bsh[2][BN * LDP];

    auto issueTile = [&](int kt, int buf) {
        int k0 = kt * BK + lhalf * 16;
        const _Float16* ga = A  + (size_t)(mBase + lrow) * FEAT + k0;
        const _Float16* gb = Bt + (size_t)(nBase + lrow) * FEAT + k0;
        _Float16* sa = &Ash[buf][lrow * LDP + lhalf * 16];
        _Float16* sb = &Bsh[buf][lrow * LDP + lhalf * 16];
        __builtin_amdgcn_global_load_async_to_lds_b128((AS_GLOBAL vi4*)ga, (AS_LDS vi4*)sa, 0,  0);
        __builtin_amdgcn_global_load_async_to_lds_b128((AS_GLOBAL vi4*)ga, (AS_LDS vi4*)sa, 16, 0);
        __builtin_amdgcn_global_load_async_to_lds_b128((AS_GLOBAL vi4*)gb, (AS_LDS vi4*)sb, 0,  0);
        __builtin_amdgcn_global_load_async_to_lds_b128((AS_GLOBAL vi4*)gb, (AS_LDS vi4*)sb, 16, 0);
    };
    auto compute = [&](int buf) {
        v16h afr[4];
#pragma unroll
        for (int i = 0; i < 4; ++i) {
            const _Float16* rp = &Ash[buf][(wm * 64 + i * 16 + lm) * LDP];
            afr[i] = frag16(rp + 8 * lh, rp + 16 + 8 * lh);
        }
        v16h bfr[2];
#pragma unroll
        for (int j = 0; j < 2; ++j) {
            const _Float16* rp = &Bsh[buf][(wn * 32 + j * 16 + lm) * LDP];
            bfr[j] = frag16(rp + 8 * lh, rp + 16 + 8 * lh);
        }
#pragma unroll
        for (int i = 0; i < 4; ++i)
#pragma unroll
            for (int j = 0; j < 2; ++j)
                acc[i][j] = __builtin_amdgcn_wmma_f32_16x16x32_f16(
                    false, afr[i], false, bfr[j], (short)0, acc[i][j], false, false);
    };

    issueTile(0, 0);
    for (int kt = 0; kt < KT - 1; ++kt) {
        __syncthreads();                               // prev compute done with buf (kt+1)&1
        issueTile(kt + 1, (kt + 1) & 1);
        __builtin_amdgcn_s_wait_asynccnt(4);           // tile kt's 4 ops complete (in-order)
        __syncthreads();                               // everyone's tile-kt data visible
        compute(kt & 1);
    }
    __syncthreads();
    __builtin_amdgcn_s_wait_asynccnt(0);
    __syncthreads();
    compute((KT - 1) & 1);
#else
    // ---- fallback: register-staged single LDS buffer ----
    __shared__ __align__(16) _Float16 Ash[BM * LDP];
    __shared__ __align__(16) _Float16 Bsh[BN * LDP];

    uint4 ra0, ra1, rb0, rb1;
    {
        int k0 = lhalf * 16;
        const uint4* pa = reinterpret_cast<const uint4*>(A  + (size_t)(mBase + lrow) * FEAT + k0);
        const uint4* pb = reinterpret_cast<const uint4*>(Bt + (size_t)(nBase + lrow) * FEAT + k0);
        ra0 = pa[0]; ra1 = pa[1];
        rb0 = pb[0]; rb1 = pb[1];
    }
    for (int kt = 0; kt < KT; ++kt) {
        __syncthreads();
        {
            uint4* sa = reinterpret_cast<uint4*>(&Ash[lrow * LDP + lhalf * 16]);
            uint4* sb = reinterpret_cast<uint4*>(&Bsh[lrow * LDP + lhalf * 16]);
            sa[0] = ra0; sa[1] = ra1;
            sb[0] = rb0; sb[1] = rb1;
        }
        __syncthreads();
        if (kt + 1 < KT) {
            int k0 = (kt + 1) * BK + lhalf * 16;
            const uint4* pa = reinterpret_cast<const uint4*>(A  + (size_t)(mBase + lrow) * FEAT + k0);
            const uint4* pb = reinterpret_cast<const uint4*>(Bt + (size_t)(nBase + lrow) * FEAT + k0);
            ra0 = pa[0]; ra1 = pa[1];
            rb0 = pb[0]; rb1 = pb[1];
        }
        v16h afr[4];
#pragma unroll
        for (int i = 0; i < 4; ++i) {
            const _Float16* rp = &Ash[(wm * 64 + i * 16 + lm) * LDP];
            afr[i] = frag16(rp + 8 * lh, rp + 16 + 8 * lh);
        }
        v16h bfr[2];
#pragma unroll
        for (int j = 0; j < 2; ++j) {
            const _Float16* rp = &Bsh[(wn * 32 + j * 16 + lm) * LDP];
            bfr[j] = frag16(rp + 8 * lh, rp + 16 + 8 * lh);
        }
#pragma unroll
        for (int i = 0; i < 4; ++i)
#pragma unroll
            for (int j = 0; j < 2; ++j)
                acc[i][j] = __builtin_amdgcn_wmma_f32_16x16x32_f16(
                    false, afr[i], false, bfr[j], (short)0, acc[i][j], false, false);
    }
#endif

    // epilogue: scores = 5*tanh(acc + bias); D layout: row = 8*lh + v, col = lm
#pragma unroll
    for (int j = 0; j < 2; ++j) {
        int col = nBase + wn * 32 + j * 16 + lm;
        float bv = bias[col];
#pragma unroll
        for (int i = 0; i < 4; ++i) {
            int row0 = mBase + wm * 64 + i * 16 + lh * 8;
            float* op = out + (size_t)row0 * COUT + col;
#pragma unroll
            for (int v = 0; v < 8; ++v)
                op[(size_t)v * COUT] = 5.0f * fast_tanh(acc[i][j][v] + bv);
        }
    }
}

// ---------------- log-semiring forward scan -> logZ[32] ---------------------
__global__ __launch_bounds__(1024)
void k_forward(const float* __restrict__ scores, float* __restrict__ logZ) {
    __shared__ float alpha[CSTATES];
    __shared__ float red[CSTATES];
    const int c = threadIdx.x;
    const int n = blockIdx.x;
    const int q = c >> 2;                    // predecessors: c, q, 256+q, 512+q, 768+q
    alpha[c] = 0.0f;
    __syncthreads();

    float s0, s1, s2, s3, s4;
    {
        const float* s = scores + (size_t)n * COUT + (size_t)c * NZ;  // t=0
        s0 = s[0]; s1 = s[1]; s2 = s[2]; s3 = s[3]; s4 = s[4];
    }
    for (int t = 0; t < T_STEPS; ++t) {
        float v0 = s0 + alpha[c];
        float v1 = s1 + alpha[q];
        float v2 = s2 + alpha[256 + q];
        float v3 = s3 + alpha[512 + q];
        float v4 = s4 + alpha[768 + q];

        float n0 = 0.f, n1 = 0.f, n2 = 0.f, n3 = 0.f, n4 = 0.f;
        if (t + 1 < T_STEPS) {               // prefetch next timestep past the chain
            const float* s = scores + (size_t)((t + 1) * NBATCH + n) * COUT + (size_t)c * NZ;
            n0 = s[0]; n1 = s[1]; n2 = s[2]; n3 = s[3]; n4 = s[4];
        }

        float m = fmaxf(fmaxf(fmaxf(v0, v1), fmaxf(v2, v3)), v4);
        float sum = __expf(v0 - m) + __expf(v1 - m) + __expf(v2 - m)
                  + __expf(v3 - m) + __expf(v4 - m);
        float an = m + __logf(sum);

        __syncthreads();
        alpha[c] = an;
        __syncthreads();
        s0 = n0; s1 = n1; s2 = n2; s3 = n3; s4 = n4;
    }

    // logsumexp over 1024 states
    red[c] = alpha[c];
    __syncthreads();
    for (int s = 512; s > 0; s >>= 1) {
        if (c < s) red[c] = fmaxf(red[c], red[c + s]);
        __syncthreads();
    }
    float m = red[0];
    __syncthreads();
    red[c] = __expf(alpha[c] - m);
    __syncthreads();
    for (int s = 512; s > 0; s >>= 1) {
        if (c < s) red[c] += red[c + s];
        __syncthreads();
    }
    if (c == 0) logZ[n] = m + __logf(red[0]);
}

// ---------------- out -= logZ[n] / T ----------------------------------------
__global__ void k_sub(float* __restrict__ out, const float* __restrict__ logZ) {
    size_t i = (size_t)blockIdx.x * 256 + threadIdx.x;
    const size_t total = (size_t)MROWS * COUT;
    if (i >= total) return;
    int n = (int)((i / COUT) & (NBATCH - 1));
    out[i] -= logZ[n] * (1.0f / (float)T_STEPS);
}

// ---------------------------------------------------------------------------
extern "C" void kernel_launch(void* const* d_in, const int* in_sizes, int n_in,
                              void* d_out, int out_size, void* d_ws, size_t ws_size,
                              hipStream_t stream) {
    const float* x      = (const float*)d_in[0];  // [32,1,4000]
    const float* conv_w = (const float*)d_in[1];  // [320,1,19]
    const float* conv_b = (const float*)d_in[2];  // [320]
    const float* lin_w  = (const float*)d_in[3];  // [320,5120]
    const float* lin_b  = (const float*)d_in[4];  // [5120]
    float* out = (float*)d_out;                   // [800,32,5120]

    char* ws = (char*)d_ws;
    _Float16* feat = (_Float16*)ws;                                  // 16,384,000 B
    _Float16* Bt   = (_Float16*)(ws + 16384000);                     //  3,276,800 B
    float*    logZ = (float*)   (ws + 16384000 + 3276800);           //        128 B

    k_wcvt<<<(FEAT * COUT + 255) / 256, 256, 0, stream>>>(lin_w, Bt);
    k_conv<<<(MROWS * FEAT + 255) / 256, 256, 0, stream>>>(x, conv_w, conv_b, feat);
    k_gemm<<<dim3(COUT / BN, MROWS / BM), 256, 0, stream>>>(feat, Bt, lin_b, out);
    k_forward<<<NBATCH, CSTATES, 0, stream>>>(out, logZ);
    k_sub<<<(unsigned)(((size_t)MROWS * COUT + 255) / 256), 256, 0, stream>>>(out, logZ);
}
